// NCP_sequence_50543175140036
// MI455X (gfx1250) — matrix operand order repaired and checked by reading, not compiled, not run on hardware
//
#include <hip/hip_runtime.h>
#include <hip/hip_bf16.h>
#include <math.h>

// ---------------------------------------------------------------------------
// NCP sequence model on MI455X (gfx1250): bf16 WMMA everywhere.
//   encoder MLP (batched GEMMs) -> per-batch-tile persistent RNN -> decoder MLP
// v2: software-pipelined K-loops (loads for kt+1 issued before WMMA kt),
//     2 N-tiles per wave in the batched GEMMs (A-fragment register reuse).
// ---------------------------------------------------------------------------

typedef __bf16 bf16t;
typedef __attribute__((ext_vector_type(16))) __bf16 v16bf;
typedef __attribute__((ext_vector_type(8)))  float  v8f;

#define B_      256
#define T_      512
#define DIN_    64
#define H1_     256
#define INTER_  128
#define STATE_  256
#define MOTOR_  64
#define DOUT_   32
#define NTOK_   (B_ * T_)          // 131072 tokens
#define HPAD_   264                // LDS row stride (bf16) -> conflict-free frag reads

// ---------------------------------------------------------------------------
// Fragment loaders (CDNA5 wave32 WMMA VGPR layouts, 16-bit data)
// A 16x32 (MxK): lane L holds row M=L%16; K base = (L>=16)?8:0;
//                element j -> K = base + (j&7) + ((j>>3)*16)
//                -> two contiguous 8-element runs => 2x b128 per lane
// B 32x16 (KxN): packed so lane L's 16 values are contiguous => 2x b128.
// ---------------------------------------------------------------------------
__device__ __forceinline__ v16bf load_a_frag(const bf16t* base, int lda, int k0) {
    int lane = threadIdx.x & 31;
    int m    = lane & 15;
    int kb   = (lane >> 4) << 3;            // 0 or 8
    const bf16t* p = base + (size_t)m * lda + k0 + kb;
    v16bf a;
#pragma unroll
    for (int v = 0; v < 8; ++v) {
        int ko = (v < 4) ? (v << 1) : (16 + ((v - 4) << 1));
        a[2 * v]     = p[ko];
        a[2 * v + 1] = p[ko + 1];
    }
    return a;
}

__device__ __forceinline__ v16bf load_b_frag(const bf16t* P, int kt, int nt, int numNT) {
    int lane = threadIdx.x & 31;
    const bf16t* p = P + (((size_t)(kt * numNT + nt)) * 32 + lane) * 16;
    v16bf b;
#pragma unroll
    for (int j = 0; j < 16; ++j) b[j] = p[j];
    return b;
}

__device__ __forceinline__ v8f wmma_bf16(v16bf a, v16bf b, v8f c) {
    return __builtin_amdgcn_wmma_f32_16x16x32_bf16(false, a, false, b, (short)0, c,
                                                   false, false);
}

// ---------------------------------------------------------------------------
// Prep kernels
// ---------------------------------------------------------------------------
__global__ void k_f32_to_bf16(const float* __restrict__ in, bf16t* __restrict__ out,
                              int n) {
    int i = blockIdx.x * blockDim.x + threadIdx.x;
    if (i < n) out[i] = (bf16t)in[i];
}

// Pack row-major W[K,N] (f32) into B-fragment order: tiles of 32x16, 32 lanes x
// 16 contiguous bf16 values per lane. lane = {kseg[1], n[3:0]}, elem j -> K=kseg*8+(j&7)+(j>>3)*16.
__global__ void k_pack_w(const float* __restrict__ W, bf16t* __restrict__ P,
                         int K, int N) {
    int i = blockIdx.x * blockDim.x + threadIdx.x;
    if (i >= K * N) return;
    int j     = i & 15;
    int lane  = (i >> 4) & 31;
    int tile  = i >> 9;
    int numNT = N >> 4;
    int nt    = tile % numNT;
    int kt    = tile / numNT;
    int n     = lane & 15;
    int kseg  = lane >> 4;
    int k     = kt * 32 + kseg * 8 + (j & 7) + ((j >> 3) << 4);
    P[i] = (bf16t)W[(size_t)k * N + nt * 16 + n];
}

// ---------------------------------------------------------------------------
// Tiled WMMA GEMM: C[M,N] = act(A[M,K] @ W[K,N] + bias)
// 256 threads = 8 waves per block; each wave owns TWO adjacent 16x16 N-tiles
// (A fragment reused by both WMMAs). K-loop software-pipelined: fragments for
// kt+1 are in flight while WMMAs for kt execute.
// ---------------------------------------------------------------------------
__global__ __launch_bounds__(256)
void k_gemm_wmma(const bf16t* __restrict__ A, const bf16t* __restrict__ P,
                 const float* __restrict__ bias, bf16t* __restrict__ Cbf,
                 float* __restrict__ Cf, int M, int N, int K, int do_tanh) {
    int wave     = threadIdx.x >> 5;
    int numNT    = N >> 4;
    int numPair  = numNT >> 1;                           // N is a multiple of 32
    int pair     = blockIdx.x * 8 + wave;                // wave-uniform
    int numPairs = (M >> 4) * numPair;
    if (pair >= numPairs) return;                        // uniform exit, EXEC full
    int pt  = pair % numPair;
    int mt  = pair / numPair;
    int nt0 = pt * 2;
    int nt1 = nt0 + 1;

    const bf16t* Abase = A + (size_t)mt * 16 * K;
    int KT = K >> 5;

    v8f acc0 = {}, acc1 = {};
    v16bf a  = load_a_frag(Abase, K, 0);
    v16bf b0 = load_b_frag(P, 0, nt0, numNT);
    v16bf b1 = load_b_frag(P, 0, nt1, numNT);
    for (int kt = 0; kt < KT; ++kt) {
        v16bf an = a, b0n = b0, b1n = b1;
        if (kt + 1 < KT) {                               // issue next-kt loads first
            an  = load_a_frag(Abase, K, (kt + 1) * 32);
            b0n = load_b_frag(P, kt + 1, nt0, numNT);
            b1n = load_b_frag(P, kt + 1, nt1, numNT);
        }
        acc0 = wmma_bf16(a, b0, acc0);
        acc1 = wmma_bf16(a, b1, acc1);
        a = an; b0 = b0n; b1 = b1n;
    }

    int lane = threadIdx.x & 31;
    int nl   = lane & 15;
    int mg   = (lane >> 4) << 3;                         // row group: 0 or 8
    int n0   = nt0 * 16 + nl;
    int n1   = nt1 * 16 + nl;
    float bv0 = bias[n0];
    float bv1 = bias[n1];
#pragma unroll
    for (int v = 0; v < 8; ++v) {
        float x0 = acc0[v] + bv0;
        float x1 = acc1[v] + bv1;
        if (do_tanh) { x0 = tanhf(x0); x1 = tanhf(x1); }
        size_t row = (size_t)(mt * 16 + mg + v);
        if (Cbf) { Cbf[row * N + n0] = (bf16t)x0; Cbf[row * N + n1] = (bf16t)x1; }
        if (Cf)  { Cf[row * N + n0]  = x0;        Cf[row * N + n1]  = x1; }
    }
}

// ---------------------------------------------------------------------------
// Persistent RNN: 16 blocks x 512 threads (16 waves). Block bb owns batch rows
// [16*bb, 16*bb+16). h kept in LDS as bf16 (padded stride). Per step:
//   all 16 waves: one state column tile, K = 128(Wih) + 256(Whh) -> 12 WMMAs
//   waves 0..3:   motor tile m = h_new @ Who + bo -> 8 WMMAs
// All fragment loads are pipelined one step ahead of their WMMA.
// ---------------------------------------------------------------------------
__global__ __launch_bounds__(512)
void k_rnn(const bf16t* __restrict__ e,        // [B*T, INTER], row = b*T + t
           const bf16t* __restrict__ WihP, const bf16t* __restrict__ WhhP,
           const float* __restrict__ bh,
           const bf16t* __restrict__ WhoP, const float* __restrict__ bo,
           bf16t* __restrict__ motors)         // [B*T, MOTOR], row = b*T + t
{
    __shared__ bf16t Hbf[16 * HPAD_];
    int wave = threadIdx.x >> 5;
    int lane = threadIdx.x & 31;
    int bb   = blockIdx.x;

    for (int i = threadIdx.x; i < 16 * HPAD_; i += blockDim.x) Hbf[i] = (bf16t)0.0f;
    __syncthreads();

    int nloc = lane & 15;
    int col  = wave * 16 + nloc;
    int mg   = (lane >> 4) << 3;
    float bhv = bh[col];
    float bov = (wave < 4) ? bo[wave * 16 + nloc] : 0.0f;

    const int KTI = INTER_ / 32;   // 4
    const int KTS = STATE_ / 32;   // 8

    for (int t = 0; t < T_; ++t) {
        const bf16t* ebase = e + ((size_t)bb * 16 * T_ + t) * INTER_;
        if (t + 1 < T_)  // warm L0/L2 for next step's encoder slice
            __builtin_prefetch(ebase + INTER_, 0, 1);

        v8f acc = {};
        v16bf a = load_a_frag(ebase, T_ * INTER_, 0);
        v16bf b = load_b_frag(WihP, 0, wave, STATE_ / 16);
#pragma unroll
        for (int kt = 0; kt < KTI; ++kt) {               // e_t @ Wih
            v16bf an, bn;
            if (kt + 1 < KTI) {
                an = load_a_frag(ebase, T_ * INTER_, (kt + 1) * 32);
                bn = load_b_frag(WihP, kt + 1, wave, STATE_ / 16);
            } else {                                     // bridge into Whh phase
                an = load_a_frag(Hbf, HPAD_, 0);
                bn = load_b_frag(WhhP, 0, wave, STATE_ / 16);
            }
            acc = wmma_bf16(a, b, acc);
            a = an; b = bn;
        }
#pragma unroll
        for (int kt = 0; kt < KTS; ++kt) {               // h @ Whh (LDS reads)
            v16bf an = a, bn = b;
            if (kt + 1 < KTS) {
                an = load_a_frag(Hbf, HPAD_, (kt + 1) * 32);
                bn = load_b_frag(WhhP, kt + 1, wave, STATE_ / 16);
            }
            acc = wmma_bf16(a, b, acc);
            a = an; b = bn;
        }
        __syncthreads();                                 // all old-h reads done
#pragma unroll
        for (int v = 0; v < 8; ++v) {
            float x = tanhf(acc[v] + bhv);
            Hbf[(mg + v) * HPAD_ + col] = (bf16t)x;
        }
        __syncthreads();                                 // new h visible

        if (wave < 4) {                                  // motor: h_new @ Who + bo
            v8f macc = {};
            v16bf ma = load_a_frag(Hbf, HPAD_, 0);
            v16bf mb = load_b_frag(WhoP, 0, wave, MOTOR_ / 16);
#pragma unroll
            for (int kt = 0; kt < KTS; ++kt) {
                v16bf man = ma, mbn = mb;
                if (kt + 1 < KTS) {
                    man = load_a_frag(Hbf, HPAD_, (kt + 1) * 32);
                    mbn = load_b_frag(WhoP, kt + 1, wave, MOTOR_ / 16);
                }
                macc = wmma_bf16(ma, mb, macc);
                ma = man; mb = mbn;
            }
#pragma unroll
            for (int v = 0; v < 8; ++v) {
                size_t row = (size_t)(bb * 16 + mg + v) * T_ + t;
                motors[row * MOTOR_ + wave * 16 + nloc] = (bf16t)(macc[v] + bov);
            }
        }
    }
}

// ---------------------------------------------------------------------------
// Host-side orchestration
// ---------------------------------------------------------------------------
extern "C" void kernel_launch(void* const* d_in, const int* in_sizes, int n_in,
                              void* d_out, int out_size, void* d_ws, size_t ws_size,
                              hipStream_t stream) {
    const float* x   = (const float*)d_in[0];
    const float* Wi1 = (const float*)d_in[1];  const float* bi1 = (const float*)d_in[2];
    const float* Wi2 = (const float*)d_in[3];  const float* bi2 = (const float*)d_in[4];
    const float* Wi3 = (const float*)d_in[5];  const float* bi3 = (const float*)d_in[6];
    const float* Wih = (const float*)d_in[7];
    const float* Whh = (const float*)d_in[8];  const float* bh  = (const float*)d_in[9];
    const float* Who = (const float*)d_in[10]; const float* bo  = (const float*)d_in[11];
    const float* Wo1 = (const float*)d_in[12]; const float* bo1 = (const float*)d_in[13];
    const float* Wo2 = (const float*)d_in[14]; const float* bo2 = (const float*)d_in[15];
    const float* Wo3 = (const float*)d_in[16]; const float* bo3 = (const float*)d_in[17];
    float* out = (float*)d_out;

    // Workspace carve-out (all buffers fully overwritten before being read).
    char* ws = (char*)d_ws;
    size_t off = 0;
    auto carve = [&](size_t bytes) -> char* {
        char* p = ws + off;
        off = (off + bytes + 255) & ~(size_t)255;
        return p;
    };
    bf16t* xbf  = (bf16t*)carve((size_t)NTOK_ * DIN_   * 2);
    bf16t* hA   = (bf16t*)carve((size_t)NTOK_ * H1_    * 2);   // ping
    bf16t* hB   = (bf16t*)carve((size_t)NTOK_ * H1_    * 2);   // pong
    bf16t* ebuf = (bf16t*)carve((size_t)NTOK_ * INTER_ * 2);
    bf16t* mot  = (bf16t*)carve((size_t)NTOK_ * MOTOR_ * 2);
    bf16t* Wi1P = (bf16t*)carve((size_t)DIN_   * H1_    * 2);
    bf16t* Wi2P = (bf16t*)carve((size_t)H1_    * H1_    * 2);
    bf16t* Wi3P = (bf16t*)carve((size_t)H1_    * INTER_ * 2);
    bf16t* WihP = (bf16t*)carve((size_t)INTER_ * STATE_ * 2);
    bf16t* WhhP = (bf16t*)carve((size_t)STATE_ * STATE_ * 2);
    bf16t* WhoP = (bf16t*)carve((size_t)STATE_ * MOTOR_ * 2);
    bf16t* Wo1P = (bf16t*)carve((size_t)MOTOR_ * H1_    * 2);
    bf16t* Wo2P = (bf16t*)carve((size_t)H1_    * H1_    * 2);
    bf16t* Wo3P = (bf16t*)carve((size_t)H1_    * DOUT_  * 2);

    // 1) x -> bf16
    {
        int n = NTOK_ * DIN_;
        k_f32_to_bf16<<<(n + 255) / 256, 256, 0, stream>>>(x, xbf, n);
    }
    // 2) pack all weights into B-fragment layout
    auto pack = [&](const float* W, bf16t* P, int K, int N) {
        int n = K * N;
        k_pack_w<<<(n + 255) / 256, 256, 0, stream>>>(W, P, K, N);
    };
    pack(Wi1, Wi1P, DIN_,   H1_);
    pack(Wi2, Wi2P, H1_,    H1_);
    pack(Wi3, Wi3P, H1_,    INTER_);
    pack(Wih, WihP, INTER_, STATE_);
    pack(Whh, WhhP, STATE_, STATE_);
    pack(Who, WhoP, STATE_, MOTOR_);
    pack(Wo1, Wo1P, MOTOR_, H1_);
    pack(Wo2, Wo2P, H1_,    H1_);
    pack(Wo3, Wo3P, H1_,    DOUT_);

    auto gemm = [&](const bf16t* A, const bf16t* P, const float* bias, bf16t* Cbf,
                    float* Cf, int N, int K, int do_tanh) {
        int pairs  = (NTOK_ / 16) * (N / 32);            // 2 N-tiles per wave
        int blocks = (pairs + 7) / 8;
        k_gemm_wmma<<<blocks, 256, 0, stream>>>(A, P, bias, Cbf, Cf, NTOK_, N, K,
                                                do_tanh);
    };

    // 3) encoder MLP
    gemm(xbf, Wi1P, bi1, hA,   nullptr, H1_,    DIN_, 1);
    gemm(hA,  Wi2P, bi2, hB,   nullptr, H1_,    H1_,  1);
    gemm(hB,  Wi3P, bi3, ebuf, nullptr, INTER_, H1_,  0);

    // 4) recurrent core: 16 persistent workgroups, one batch tile each
    k_rnn<<<B_ / 16, 512, 0, stream>>>(ebuf, WihP, WhhP, bh, WhoP, bo, mot);

    // 5) decoder MLP (final layer writes f32 directly to d_out, rows = b*T + t)
    gemm(mot, Wo1P, bo1, hA, nullptr, H1_,   MOTOR_, 1);
    gemm(hA,  Wo2P, bo2, hB, nullptr, H1_,   H1_,    1);
    gemm(hB,  Wo3P, bo3, nullptr, out, DOUT_, H1_,   0);
}
